// SkipIPAGNNSingle_87935160418877
// MI455X (gfx1250) — matrix operand order repaired and checked by reading, not compiled, not run on hardware
//
#include <hip/hip_runtime.h>

// ---------------------------------------------------------------------------
// SkipIPAGNN on gfx1250: f16 WMMA for the big GEMMs, fused skip-LSTM + IPA
// reduction, async global->LDS double-buffered staging in the hot loop,
// deterministic gather-based aggregation (no float atomics).
// N=256 nodes, H=128 hidden, 4H=512 gates, V=64 vocab, 4 steps.
// ---------------------------------------------------------------------------

typedef __attribute__((ext_vector_type(16))) _Float16 v16h;
typedef __attribute__((ext_vector_type(8)))  _Float16 v8h;
typedef __attribute__((ext_vector_type(8)))  float    v8f;

// 16-byte vector type exactly as the async-to-LDS builtin expects it
// (per hipcc diagnostic: '__vector_size__(4*sizeof(int)) int __device__ *').
typedef int v4i_ __attribute__((__vector_size__(16)));
typedef __attribute__((address_space(1))) v4i_* gbl128_t;
typedef __attribute__((address_space(3))) v4i_* lds128_t;

#define NN   256
#define HH   128
#define G4   512

#if __has_builtin(__builtin_amdgcn_global_load_async_to_lds_b128) && \
    __has_builtin(__builtin_amdgcn_s_wait_asynccnt)
#define USE_ASYNC 1
#else
#define USE_ASYNC 0
#endif

__device__ __forceinline__ float sigf(float x) {
    return 1.0f / (1.0f + __expf(-x));
}
__device__ __forceinline__ float tanh_f(float x) {
    float a = fabsf(x);
    float e = __expf(-2.0f * a);
    float t = (1.0f - e) / (1.0f + e);
    return (x < 0.0f) ? -t : t;
}

// Stage 32 bytes (8 consecutive floats) global -> LDS, asynchronously if the
// CDNA5 async-to-LDS path is available (tracked by ASYNCcnt).
__device__ __forceinline__ void stage32(const float* __restrict__ g, float* l) {
#if USE_ASYNC
    __builtin_amdgcn_global_load_async_to_lds_b128(
        (gbl128_t)(g),     (lds128_t)(l),     0, 0);
    __builtin_amdgcn_global_load_async_to_lds_b128(
        (gbl128_t)(g + 4), (lds128_t)(l + 4), 0, 0);
#else
    float4 a = *(const float4*)g;
    float4 b = *(const float4*)(g + 4);
    *(float4*)l       = a;
    *(float4*)(l + 4) = b;
#endif
}
__device__ __forceinline__ void wait_async_le2() {
#if USE_ASYNC
    __builtin_amdgcn_s_wait_asynccnt(2);
#endif
}

// A-fragment (16x32 f16, M=rows, K=32*kk..) built from an f16 LDS tile.
// Per ISA table: lane L<16: M=L, K={0..7,16..23}; lane L>=16: K={8..15,24..31}.
__device__ __forceinline__ v16h load_a_frag(const _Float16* lds, int lane, int kk) {
    int jm = lane & 15;
    int kb = (lane < 16) ? 0 : 8;
    const v8h* p0 = (const v8h*)(lds + jm * HH + kk * 32 + kb);
    const v8h* p1 = (const v8h*)(lds + jm * HH + kk * 32 + kb + 16);
    v8h a = *p0, b = *p1;
    v16h r;
#pragma unroll
    for (int e = 0; e < 8; ++e) { r[e] = a[e]; r[e + 8] = b[e]; }
    return r;
}

// Same A-fragment built from an f32 LDS tile (convert at fragment build).
__device__ __forceinline__ v16h load_a_frag_f32(const float* lds, int lane, int kk) {
    int jm = lane & 15;
    int kb = (lane < 16) ? 0 : 8;
    const float* p0 = lds + jm * HH + kk * 32 + kb;
    const float* p1 = p0 + 16;
    v16h r;
#pragma unroll
    for (int e = 0; e < 8; ++e) {
        r[e]     = (_Float16)p0[e];
        r[e + 8] = (_Float16)p1[e];
    }
    return r;
}

// B-fragment (32x16 f16) from row-major f32 B[k*ldb + n]:
// lane L: N = n0 + (L&15); K = 32*kk + ((L<16)?0:8) + (e<8 ? e : e+8)
__device__ __forceinline__ v16h load_b_frag(const float* __restrict__ B, int ldb,
                                            int n0, int kk, int lane) {
    int n  = n0 + (lane & 15);
    int kb = (lane < 16) ? 0 : 8;
    v16h r;
#pragma unroll
    for (int e = 0; e < 16; ++e)
        r[e] = (_Float16)B[(kk * 32 + kb + (e < 8 ? e : e + 8)) * ldb + n];
    return r;
}

#define WMMA(a, b, c) \
    __builtin_amdgcn_wmma_f32_16x16x32_f16(false, (a), false, (b), (short)0, (c), false, false)

// ---------------------------------------------------------------------------
// init: c = h = 0, ip = e0
__global__ __launch_bounds__(128) void k_init(float* c, float* h, float* ip) {
    int j = blockIdx.x, t = threadIdx.x;
    c[j * HH + t] = 0.0f;
    h[j * HH + t] = 0.0f;
    if (t == 0) ip[j] = (j == 0) ? 1.0f : 0.0f;
}

// xWx = node_emb @ Wx + b   [256,512]  (one-time)
__global__ __launch_bounds__(128) void k_xwx(const float* __restrict__ ne,
                                             const float* __restrict__ Wx,
                                             const float* __restrict__ b,
                                             float* __restrict__ xWx) {
    int i = blockIdx.x, t = threadIdx.x;
    __shared__ float x[HH];
    x[t] = ne[i * HH + t];
    __syncthreads();
#pragma unroll
    for (int g = 0; g < 4; ++g) {
        int col = g * HH + t;
        float s = b[col];
        for (int k = 0; k < HH; ++k) s += x[k] * Wx[k * G4 + col];
        xWx[i * G4 + col] = s;
    }
}

// keysWl[i,j] = sum_k relu(skip_emb[i,j]@Wk + bk)[k] * Wl[k]   (one-time)
__global__ __launch_bounds__(128) void k_keys(const float* __restrict__ se,
                                              const float* __restrict__ Wk,
                                              const float* __restrict__ bk,
                                              const float* __restrict__ Wl,
                                              float* __restrict__ keysWl) {
    int ij = blockIdx.x, t = threadIdx.x;
    __shared__ float x[HH];
    __shared__ float red[HH];
    x[t] = se[(size_t)ij * HH + t];
    __syncthreads();
    float s = bk[t];
    for (int m = 0; m < HH; ++m) s += x[m] * Wk[m * HH + t];
    red[t] = fmaxf(s, 0.0f) * Wl[t];
    __syncthreads();
    for (int o = 64; o > 0; o >>= 1) {
        if (t < o) red[t] += red[t + o];
        __syncthreads();
    }
    if (t == 0) keysWl[ij] = red[0];
}

// ---------------------------------------------------------------------------
// Per-step node kernel: gates = xWx + h@Wh (WMMA) -> c_p,h_p; hWhs = h@Whs + bs
// (WMMA); queries->relu->qWl; branch softmax.  16 blocks x 256 threads.
__global__ __launch_bounds__(256) void k_node(
    const float* __restrict__ xWx, const float* __restrict__ Wh,
    const float* __restrict__ Whs, const float* __restrict__ bs,
    const float* __restrict__ Wq, const float* __restrict__ bq,
    const float* __restrict__ Wb, const float* __restrict__ bb,
    const float* __restrict__ Wl, const float* __restrict__ bl,
    const float* __restrict__ c, const float* __restrict__ h,
    float* __restrict__ cp, float* __restrict__ hp,
    float* __restrict__ hWhs, float* __restrict__ branch,
    float* __restrict__ qWl) {
    __shared__ __align__(16) _Float16 hA[16 * HH];
    __shared__ float hid[16 * 256];
    __shared__ float qrelu[16 * HH];
    const int tid = threadIdx.x, lane = tid & 31, wv = tid >> 5;
    const int i0 = blockIdx.x * 16;

    for (int t = tid; t < 16 * HH; t += 256) {
        int r = t >> 7, k = t & 127;
        hA[t] = (_Float16)h[(i0 + r) * HH + k];
    }
    __syncthreads();

    v16h afr[4];
#pragma unroll
    for (int kk = 0; kk < 4; ++kk) afr[kk] = load_a_frag(hA, lane, kk);

    const int hc0 = wv * 16, nIdx = lane & 15, rowOff = (lane >> 4) * 8;
    const int hcol = hc0 + nIdx;
    const v8f z = {0.f, 0.f, 0.f, 0.f, 0.f, 0.f, 0.f, 0.f};
    v8f accN[4], accS[4];
#pragma unroll
    for (int g = 0; g < 4; ++g) {
        v8f aN = z, aS = z;
#pragma unroll
        for (int kk = 0; kk < 4; ++kk) {
            v16h bN = load_b_frag(Wh, G4, hc0 + 128 * g, kk, lane);
            aN = WMMA(afr[kk], bN, aN);
            v16h bS = load_b_frag(Whs, G4, hc0 + 128 * g, kk, lane);
            aS = WMMA(afr[kk], bS, aS);
        }
        accN[g] = aN; accS[g] = aS;
    }
    // store h@Whs + bs for the skip kernel
#pragma unroll
    for (int g = 0; g < 4; ++g) {
        int col = hc0 + 128 * g + nIdx;
        float bsv = bs[col];
#pragma unroll
        for (int r = 0; r < 8; ++r) {
            int i = i0 + rowOff + r;
            hWhs[i * G4 + col] = accS[g][r] + bsv;
        }
    }
    // per-node LSTM
#pragma unroll
    for (int r = 0; r < 8; ++r) {
        int row = rowOff + r, i = i0 + row;
        float gi = accN[0][r] + xWx[i * G4 + hcol];
        float gf = accN[1][r] + xWx[i * G4 + 128 + hcol];
        float gg = accN[2][r] + xWx[i * G4 + 256 + hcol];
        float go = accN[3][r] + xWx[i * G4 + 384 + hcol];
        float cv = c[i * HH + hcol];
        float c2 = sigf(gf) * cv + sigf(gi) * tanh_f(gg);
        float h2 = sigf(go) * tanh_f(c2);
        cp[i * HH + hcol] = c2;
        hp[i * HH + hcol] = h2;
        hid[row * 256 + hcol]       = c2;
        hid[row * 256 + 128 + hcol] = h2;
    }
    __syncthreads();
    // queries = relu(hid @ Wq + bq)
    for (int t = tid; t < 16 * HH; t += 256) {
        int r = t >> 7, k = t & 127;
        float s = bq[k];
        for (int m = 0; m < 256; ++m) s += hid[r * 256 + m] * Wq[m * HH + k];
        qrelu[t] = fmaxf(s, 0.0f);
    }
    __syncthreads();
    if (tid < 16) {
        int r = tid, i = i0 + r;
        float s = bl[0];
        for (int k = 0; k < HH; ++k) s += qrelu[r * HH + k] * Wl[HH + k];
        qWl[i] = s;
        float l0 = bb[0], l1 = bb[1];
        for (int m = 0; m < 256; ++m) {
            float hv = hid[r * 256 + m];
            l0 += hv * Wb[m * 2];
            l1 += hv * Wb[m * 2 + 1];
        }
        float mx = fmaxf(l0, l1);
        float e0 = __expf(l0 - mx), e1 = __expf(l1 - mx);
        float inv = 1.0f / (e0 + e1);
        branch[i * 2]     = e0 * inv;
        branch[i * 2 + 1] = e1 * inv;
    }
}

// Row-wise masked softmax; W[i,j] = ip[i]*yes_skip[i,j]; no_skip[i] = diag.
__global__ __launch_bounds__(128) void k_softmax(const float* __restrict__ keysWl,
                                                 const float* __restrict__ qWl,
                                                 const float* __restrict__ ip,
                                                 float* __restrict__ W,
                                                 float* __restrict__ no_skip) {
    int i = blockIdx.x, t = threadIdx.x;
    __shared__ float red[128];
    float q = qWl[i];
    float l0 = (t >= i)       ? keysWl[i * NN + t] + q       : -3.0e38f;
    float l1 = (t + 128 >= i) ? keysWl[i * NN + t + 128] + q : -3.0e38f;
    red[t] = fmaxf(l0, l1);
    __syncthreads();
    for (int o = 64; o > 0; o >>= 1) {
        if (t < o) red[t] = fmaxf(red[t], red[t + o]);
        __syncthreads();
    }
    float m = red[0];
    __syncthreads();
    float e0 = (t >= i)       ? __expf(l0 - m) : 0.0f;
    float e1 = (t + 128 >= i) ? __expf(l1 - m) : 0.0f;
    red[t] = e0 + e1;
    __syncthreads();
    for (int o = 64; o > 0; o >>= 1) {
        if (t < o) red[t] += red[t + o];
        __syncthreads();
    }
    float inv = 1.0f / red[0];
    float d0 = e0 * inv, d1 = e1 * inv;
    if (t == i)       no_skip[i] = d0;
    if (t + 128 == i) no_skip[i] = d1;
    float ipi = ip[i];
    W[i * NN + t]       = ipi * ((t == i)       ? 0.0f : d0);
    W[i * NN + t + 128] = ipi * ((t + 128 == i) ? 0.0f : d1);
}

// new_ip[j] = col-sum of W + gathered branch mass (deterministic, no atomics)
__global__ __launch_bounds__(256) void k_newip(const float* __restrict__ W,
                                               const float* __restrict__ ip,
                                               const float* __restrict__ ns,
                                               const float* __restrict__ branch,
                                               const int* __restrict__ ti,
                                               const int* __restrict__ fi,
                                               float* __restrict__ new_ip) {
    int j = threadIdx.x;
    float s = 0.0f;
    for (int i = 0; i < NN; ++i) {
        s += W[i * NN + j];
        float p = ip[i] * ns[i];
        if (ti[i] == j) s += p * branch[i * 2];
        if (fi[i] == j) s += p * branch[i * 2 + 1];
    }
    new_ip[j] = s;
}

// ---------------------------------------------------------------------------
// Fused skip-LSTM + weighted reduction over sources i, with async
// double-buffered A-tile staging (ASYNCcnt-tracked global->LDS B128 copies).
// Block b owns j in [16b,16b+16); wave w owns hcols [16w,16w+16).
__global__ __launch_bounds__(256) void k_skip(
    const float* __restrict__ skip_emb, const float* __restrict__ Wxs,
    const float* __restrict__ hWhs, const float* __restrict__ c,
    const float* __restrict__ W,
    float* __restrict__ acc_c, float* __restrict__ acc_h) {
    __shared__ __align__(16) float At[2][16 * HH];
    const int tid = threadIdx.x, lane = tid & 31, wv = tid >> 5;
    const int j0 = blockIdx.x * 16;
    const int hc0 = wv * 16, nIdx = lane & 15, rowOff = (lane >> 4) * 8;
    const int hcol = hc0 + nIdx;

    // register-resident B fragments of Wxs (i-invariant)
    v16h bfr[4][4];
#pragma unroll
    for (int g = 0; g < 4; ++g)
#pragma unroll
        for (int kk = 0; kk < 4; ++kk)
            bfr[g][kk] = load_b_frag(Wxs, G4, hc0 + 128 * g, kk, lane);

    // per-thread 32B staging slice of the 16x128 f32 tile
    const int sr  = tid >> 4;        // row 0..15
    const int skc = (tid & 15) * 8;  // col 0,8,...,120
    const float* gsrc = skip_emb + (j0 + sr) * HH + skc;
    float* l0 = &At[0][sr * HH + skc];
    float* l1 = &At[1][sr * HH + skc];

    stage32(gsrc, l0);  // tile 0 in flight

    float sc[8], sh[8];
#pragma unroll
    for (int r = 0; r < 8; ++r) { sc[r] = 0.0f; sh[r] = 0.0f; }
    const v8f z = {0.f, 0.f, 0.f, 0.f, 0.f, 0.f, 0.f, 0.f};

    for (int i = 0; i < NN; ++i) {
        __syncthreads();  // everyone done reading the buffer we now overwrite
        int inext = (i + 1 < NN) ? (i + 1) : (NN - 1);
        stage32(gsrc + (size_t)inext * NN * HH, ((i + 1) & 1) ? l1 : l0);
        wait_async_le2();  // in-order: tile i's two B128s retired
        __syncthreads();   // whole tile i resident for all waves

        const float* Acur = At[i & 1];
        v16h afr[4];
#pragma unroll
        for (int kk = 0; kk < 4; ++kk) afr[kk] = load_a_frag_f32(Acur, lane, kk);
        v8f ga = z, gb = z, gc = z, gd = z;
#pragma unroll
        for (int kk = 0; kk < 4; ++kk) {
            ga = WMMA(afr[kk], bfr[0][kk], ga);
            gb = WMMA(afr[kk], bfr[1][kk], gb);
            gc = WMMA(afr[kk], bfr[2][kk], gc);
            gd = WMMA(afr[kk], bfr[3][kk], gd);
        }
        float h0 = hWhs[i * G4 + hcol];
        float h1 = hWhs[i * G4 + 128 + hcol];
        float h2 = hWhs[i * G4 + 256 + hcol];
        float h3 = hWhs[i * G4 + 384 + hcol];
        float cv = c[i * HH + hcol];
#pragma unroll
        for (int r = 0; r < 8; ++r) {
            float gi = ga[r] + h0, gf = gb[r] + h1;
            float gg = gc[r] + h2, go = gd[r] + h3;
            float c2 = sigf(gf) * cv + sigf(gi) * tanh_f(gg);
            float hv = sigf(go) * tanh_f(c2);
            float wgt = W[i * NN + j0 + rowOff + r];
            sc[r] += wgt * c2;
            sh[r] += wgt * hv;
        }
    }
#pragma unroll
    for (int r = 0; r < 8; ++r) {
        int j = j0 + rowOff + r;
        acc_c[j * HH + hcol] = sc[r];
        acc_h[j * HH + hcol] = sh[r];
    }
}

// combine: gather branch contributions (deterministic) + skip part, /denom.
__global__ __launch_bounds__(128) void k_combine(
    const float* __restrict__ cp, const float* __restrict__ hp,
    const float* __restrict__ ip, const float* __restrict__ ns,
    const float* __restrict__ branch, const int* __restrict__ ti,
    const int* __restrict__ fi, const float* __restrict__ acc_c,
    const float* __restrict__ acc_h, const float* __restrict__ new_ip,
    float* __restrict__ c, float* __restrict__ h) {
    int j = blockIdx.x, t = threadIdx.x;
    float sc = acc_c[j * HH + t];
    float sh = acc_h[j * HH + t];
    for (int i = 0; i < NN; ++i) {
        float p = ip[i] * ns[i];
        float w = 0.0f;
        if (ti[i] == j) w += p * branch[i * 2];
        if (fi[i] == j) w += p * branch[i * 2 + 1];
        if (w != 0.0f) {
            sc += w * cp[i * HH + t];
            sh += w * hp[i * HH + t];
        }
    }
    float d = new_ip[j] + 1e-7f;
    c[j * HH + t] = sc / d;
    h[j * HH + t] = sh / d;
}

__global__ __launch_bounds__(256) void k_cpip(const float* __restrict__ nip,
                                              float* __restrict__ ip) {
    ip[threadIdx.x] = nip[threadIdx.x];
}

// out = concat(c[ei], h[ei]) @ Wd + bd
__global__ __launch_bounds__(64) void k_out(const float* __restrict__ c,
                                            const float* __restrict__ h,
                                            const float* __restrict__ Wd,
                                            const float* __restrict__ bd,
                                            const int* __restrict__ exit_idx,
                                            float* __restrict__ out) {
    int v = threadIdx.x;
    int ei = exit_idx[0];
    float s = bd[v];
    for (int k = 0; k < HH; ++k) s += c[ei * HH + k] * Wd[k * 64 + v];
    for (int k = 0; k < HH; ++k) s += h[ei * HH + k] * Wd[(HH + k) * 64 + v];
    out[v] = s;
}

// ---------------------------------------------------------------------------
extern "C" void kernel_launch(void* const* d_in, const int* in_sizes, int n_in,
                              void* d_out, int out_size, void* d_ws, size_t ws_size,
                              hipStream_t stream) {
    (void)in_sizes; (void)n_in; (void)out_size; (void)ws_size;
    const float* node_emb = (const float*)d_in[0];
    const float* skip_emb = (const float*)d_in[1];
    const float* Wx  = (const float*)d_in[2];
    const float* Wh  = (const float*)d_in[3];
    const float* b   = (const float*)d_in[4];
    const float* Wxs = (const float*)d_in[5];
    const float* Whs = (const float*)d_in[6];
    const float* bs  = (const float*)d_in[7];
    const float* Wb  = (const float*)d_in[8];
    const float* bb  = (const float*)d_in[9];
    const float* Wk  = (const float*)d_in[10];
    const float* bk  = (const float*)d_in[11];
    const float* Wq  = (const float*)d_in[12];
    const float* bq  = (const float*)d_in[13];
    const float* Wl  = (const float*)d_in[14];
    const float* bl  = (const float*)d_in[15];
    const float* Wd  = (const float*)d_in[16];
    const float* bd  = (const float*)d_in[17];
    // d_in[18] = skip_mask (triangular; recomputed on device as j >= i)
    const int* ti = (const int*)d_in[19];
    const int* fi = (const int*)d_in[20];
    const int* ei = (const int*)d_in[21];
    float* out = (float*)d_out;

    float* w = (float*)d_ws;
    float* xWx    = w; w += 256 * 512;
    float* keysWl = w; w += 256 * 256;
    float* C      = w; w += 256 * 128;
    float* Hh     = w; w += 256 * 128;
    float* IP     = w; w += 256;
    float* CP     = w; w += 256 * 128;
    float* HP     = w; w += 256 * 128;
    float* HWHS   = w; w += 256 * 512;
    float* BR     = w; w += 512;
    float* QW     = w; w += 256;
    float* WIJ    = w; w += 256 * 256;
    float* NS     = w; w += 256;
    float* NIP    = w; w += 256;
    float* AC     = w; w += 256 * 128;
    float* AH     = w; w += 256 * 128;

    k_init<<<256, 128, 0, stream>>>(C, Hh, IP);
    k_xwx<<<256, 128, 0, stream>>>(node_emb, Wx, b, xWx);
    k_keys<<<65536, 128, 0, stream>>>(skip_emb, Wk, bk, Wl, keysWl);

    for (int s = 0; s < 4; ++s) {
        k_node<<<16, 256, 0, stream>>>(xWx, Wh, Whs, bs, Wq, bq, Wb, bb, Wl, bl,
                                       C, Hh, CP, HP, HWHS, BR, QW);
        k_softmax<<<256, 128, 0, stream>>>(keysWl, QW, IP, WIJ, NS);
        k_newip<<<1, 256, 0, stream>>>(WIJ, IP, NS, BR, ti, fi, NIP);
        k_skip<<<16, 256, 0, stream>>>(skip_emb, Wxs, HWHS, C, WIJ, AC, AH);
        k_combine<<<256, 128, 0, stream>>>(CP, HP, IP, NS, BR, ti, fi, AC, AH,
                                           NIP, C, Hh);
        k_cpip<<<1, 256, 0, stream>>>(NIP, IP);
    }
    k_out<<<1, 64, 0, stream>>>(C, Hh, Wd, bd, ei, out);
}
